// _ScratchXLSTM_32770600468731
// MI455X (gfx1250) — matrix operand order repaired and checked by reading, not compiled, not run on hardware
//
#include <hip/hip_runtime.h>
#include <hip/hip_bf16.h>

#define EE   128   // embed dim
#define SS   64    // seq len
#define BB   128   // batch
#define HH   4     // heads
#define DD   32    // head dim
#define VV   8192  // vocab
#define NBLK 2
#define ROWS (BB*SS)   // 8192
#define NTB  (BB/16)   // 8 batch tiles
#define EPSV 1e-5f

typedef __attribute__((ext_vector_type(16))) __bf16 v16bf;
typedef __attribute__((ext_vector_type(8)))  float  v8f;

union FragAB { v16bf v; unsigned u[8]; };
union FragC  { v8f v;  float f[8]; };

__device__ __forceinline__ unsigned short f2bf(float f) {
    unsigned u = __float_as_uint(f);
    u += 0x7fffu + ((u >> 16) & 1u);   // round-to-nearest-even
    return (unsigned short)(u >> 16);
}

__device__ __forceinline__ void frag_from_2q(FragAB& fr, uint4 q0, uint4 q1) {
    fr.u[0] = q0.x; fr.u[1] = q0.y; fr.u[2] = q0.z; fr.u[3] = q0.w;
    fr.u[4] = q1.x; fr.u[5] = q1.y; fr.u[6] = q1.z; fr.u[7] = q1.w;
}

// ---------------- weight conversion ----------------
__global__ void k_f32_to_bf16(const float* __restrict__ src, unsigned short* __restrict__ dst, int n) {
    int i = blockIdx.x * 256 + threadIdx.x;
    if (i < n) dst[i] = f2bf(src[i]);
}

// src[k*N+n] (K x N) -> dst[n*K+k] (N x K): B-fragment K-pairs contiguous
__global__ void k_f32_to_bf16_tr(const float* __restrict__ src, unsigned short* __restrict__ dst, int K, int N) {
    int i = blockIdx.x * 256 + threadIdx.x;
    if (i < K * N) { int k = i / N, n = i % N; dst[n * K + k] = f2bf(src[i]); }
}

// ---------------- embedding gather ----------------
__global__ void k_embed(const int* __restrict__ x, const float* __restrict__ emb, float* __restrict__ h) {
    int row = blockIdx.x, t = threadIdx.x;
    h[row * EE + t] = emb[x[row] * EE + t];
}

// ---------------- layernorm (E=128) -> bf16 ----------------
__global__ void k_layernorm_bf(const float* __restrict__ in, const float* __restrict__ s,
                               const float* __restrict__ b, unsigned short* __restrict__ out) {
    __shared__ float red[EE];
    int row = blockIdx.x, t = threadIdx.x;
    float v = in[row * EE + t];
    red[t] = v; __syncthreads();
    for (int off = 64; off > 0; off >>= 1) { if (t < off) red[t] += red[t + off]; __syncthreads(); }
    float mu = red[0] * (1.f / EE); __syncthreads();
    float d = v - mu;
    red[t] = d * d; __syncthreads();
    for (int off = 64; off > 0; off >>= 1) { if (t < off) red[t] += red[t + off]; __syncthreads(); }
    float var = red[0] * (1.f / EE);
    float y = d * rsqrtf(var + EPSV) * s[t] + b[t];
    out[row * EE + t] = f2bf(y);
}

// ---------------- gate pre-projection gx = xln @ W + bg ----------------
// grid (S, NTB); block 128: wave = gate g.
// gx stored FRAGMENT-MAJOR: fragIdx = ((((s*4+g)*HH+h)*NTB+tb)*2+nt), each frag = 32 lanes x 8 f32.
__global__ void k_gx(const unsigned short* __restrict__ xln, const unsigned short* __restrict__ Wbf,
                     const float* __restrict__ bg, float* __restrict__ gx) {
    int s  = blockIdx.x;
    int tb = blockIdx.y;
    int b0 = tb * 16;
    int l = threadIdx.x & 31, g = threadIdx.x >> 5;
    int n = l & 15, half = l >> 4;
    for (int h = 0; h < HH; ++h) {
        FragAB a;
        {
            int rowA = (b0 + n) * SS + s;          // A: M = l&15
            const uint4* p = (const uint4*)(xln + rowA * EE + h * DD + (half << 4));
            frag_from_2q(a, p[0], p[1]);           // 16 contiguous ushorts
        }
        #pragma unroll
        for (int nt = 0; nt < 2; ++nt) {
            int ocol = nt * 16 + n;
            FragAB bf;
            {
                const uint4* p = (const uint4*)(Wbf + ((g * HH + h) * DD + ocol) * DD + (half << 4));
                frag_from_2q(bf, p[0], p[1]);
            }
            FragC c;
            float bias = bg[(g * HH + h) * DD + ocol];
            #pragma unroll
            for (int r = 0; r < 8; ++r) c.f[r] = bias;
            c.v = __builtin_amdgcn_wmma_f32_16x16x32_bf16(false, a.v, false, bf.v, (short)0, c.v, false, false);
            long fragIdx = ((((long)(s * 4 + g) * HH + h) * NTB + tb) * 2 + nt);
            *(v8f*)(gx + fragIdx * 256 + l * 8) = c.v;   // contiguous 32B per lane
        }
    }
}

// ---------------- sLSTM sequential scan ----------------
// grid (NTB, HH); block 32 = ONE wave per (batch-tile, head). No barriers needed.
// y written FRAGMENT-MAJOR: base = (((tb*HH+w)*2+nt)*SS+s)*256 + l*8.
__global__ void k_scan(const float* __restrict__ gx, const unsigned short* __restrict__ Rbf,
                       float* __restrict__ y) {
    __shared__ __align__(16) unsigned short hlds[16 * DD];   // h_prev tile, bf16 (1KB)
    int tb = blockIdx.x;
    int w  = blockIdx.y;                       // head
    int l = threadIdx.x & 31;
    int n = l & 15, half = l >> 4;
    for (int i = threadIdx.x; i < 16 * DD; i += 32) hlds[i] = 0;

    // preload recurrent weight fragments R[g][head w] (32x32) as 2 N-tiles each
    FragAB rf[4][2];
    #pragma unroll
    for (int g = 0; g < 4; ++g)
      #pragma unroll
      for (int nt = 0; nt < 2; ++nt) {
          const uint4* p = (const uint4*)(Rbf + ((g * HH + w) * DD + nt * 16 + n) * DD + (half << 4));
          frag_from_2q(rf[g][nt], p[0], p[1]);
      }

    FragC cst[2], nst[2], mst[2];
    #pragma unroll
    for (int nt = 0; nt < 2; ++nt)
      #pragma unroll
      for (int r = 0; r < 8; ++r) { cst[nt].f[r] = 0.f; nst[nt].f[r] = 0.f; mst[nt].f[r] = -__builtin_inff(); }

    for (int s = 0; s < SS; ++s) {
        FragAB a;   // A = h_prev[16 rows][this head's 32 cols], from LDS
        {
            const uint4* p = (const uint4*)&hlds[n * DD + (half << 4)];
            frag_from_2q(a, p[0], p[1]);   // 16 contiguous ushorts
        }
        FragC pre[4][2];
        #pragma unroll
        for (int g = 0; g < 4; ++g)
          #pragma unroll
          for (int nt = 0; nt < 2; ++nt) {
            long fragIdx = ((((long)(s * 4 + g) * HH + w) * NTB + tb) * 2 + nt);
            FragC c;
            c.v = *(const v8f*)(gx + fragIdx * 256 + l * 8);
            pre[g][nt].v = __builtin_amdgcn_wmma_f32_16x16x32_bf16(false, a.v, false, rf[g][nt].v, (short)0, c.v, false, false);
          }
        // prefetch next step's gx fragments into WGP cache while gating math runs
        if (s + 1 < SS) {
            #pragma unroll
            for (int g = 0; g < 4; ++g)
              #pragma unroll
              for (int nt = 0; nt < 2; ++nt) {
                long fragIdx = ((((long)((s + 1) * 4 + g) * HH + w) * NTB + tb) * 2 + nt);
                __builtin_prefetch(gx + fragIdx * 256 + l * 8, 0, 3);
              }
        }
        #pragma unroll
        for (int nt = 0; nt < 2; ++nt) {
          FragC hv;
          #pragma unroll
          for (int r = 0; r < 8; ++r) {
            float iraw = pre[0][nt].f[r], fraw = pre[1][nt].f[r];
            float zraw = pre[2][nt].f[r], oraw = pre[3][nt].f[r];
            float ls   = fminf(fraw, 0.f) - log1pf(__expf(-fabsf(fraw)));   // log_sigmoid
            float lgf  = mst[nt].f[r] + ls;
            float mnew = fmaxf(iraw, lgf);
            float ig = __expf(iraw - mnew);
            float fg = __expf(lgf - mnew);
            float cnew = fg * cst[nt].f[r] + ig * tanhf(zraw);
            float nnew = fg * nst[nt].f[r] + ig;
            float hnew = (1.f / (1.f + __expf(-oraw))) * cnew / nnew;
            cst[nt].f[r] = cnew; nst[nt].f[r] = nnew; mst[nt].f[r] = mnew;
            hv.f[r] = hnew;
            hlds[(r + (half << 3)) * DD + nt * 16 + n] = f2bf(hnew);
          }
          long base = (((long)(tb * HH + w) * 2 + nt) * SS + s) * 256;
          *(v8f*)(y + base + l * 8) = hv.v;    // 2x b128 per nt
        }
    }
}

// ---------------- fused: h += headnorm(y); z_bf = layernorm(h, ln2) ----------------
// y is fragment-major (see k_scan).
__global__ void k_hn_res_ln(const float* __restrict__ y, const float* __restrict__ gs,
                            const float* __restrict__ gb, const float* __restrict__ ls,
                            const float* __restrict__ lb, float* __restrict__ h,
                            unsigned short* __restrict__ zbf) {
    __shared__ float red[EE];
    int row = blockIdx.x, t = threadIdx.x;   // head = t/32 == wave
    int b = row >> 6, s = row & (SS - 1);
    int tb = b >> 4, m = b & 15;
    int wh = t >> 5, nt = (t >> 4) & 1, n = t & 15;
    int l = n + ((m >> 3) << 4);
    int r = m & 7;
    float v = y[(((long)(tb * HH + wh) * 2 + nt) * SS + s) * 256 + l * 8 + r];
    float sum = v;
    for (int off = 16; off > 0; off >>= 1) sum += __shfl_xor(sum, off, 32);
    float mu = sum * (1.f / DD);
    float d = v - mu;
    float sq = d * d;
    for (int off = 16; off > 0; off >>= 1) sq += __shfl_xor(sq, off, 32);
    float var = sq * (1.f / DD);
    float yn = d * rsqrtf(var + EPSV);
    float hn = h[row * EE + t] + yn * gs[t] + gb[t];
    h[row * EE + t] = hn;
    // layernorm over full E=128 row
    red[t] = hn; __syncthreads();
    for (int off = 64; off > 0; off >>= 1) { if (t < off) red[t] += red[t + off]; __syncthreads(); }
    float mu2 = red[0] * (1.f / EE); __syncthreads();
    float d2 = hn - mu2;
    red[t] = d2 * d2; __syncthreads();
    for (int off = 64; off > 0; off >>= 1) { if (t < off) red[t] += red[t + off]; __syncthreads(); }
    float var2 = red[0] * (1.f / EE);
    zbf[row * EE + t] = f2bf(d2 * rsqrtf(var2 + EPSV) * ls[t] + lb[t]);
}

// ---------------- GEMM  out = A(Mx128 bf16) @ B(128xN bf16, stored NxK) + bias ----------------
// mode 0: outf = val; mode 1: outbf = bf16(gelu(val)); mode 2: outf = res + val
// grid (M/64, N/128); block 128. 64x128 output tile: A staged in LDS (shared by all 4 waves),
// B fragments reused across 4 M-tiles, output staged in LDS for coalesced b128 stores.
__global__ void k_gemm128(const unsigned short* __restrict__ A, const unsigned short* __restrict__ Bt,
                          const float* __restrict__ bias, const float* __restrict__ res,
                          float* __restrict__ outf, unsigned short* __restrict__ outbf,
                          int N, int mode) {
    __shared__ __align__(16) char smem[64 * EE * 4];   // 32KB: A stage (16KB), then out stage
    int tid = threadIdx.x;
    int m0 = blockIdx.x * 64;
    int ncolBase = blockIdx.y * 128;
    int l = tid & 31, w = tid >> 5;
    int n = l & 15, half = l >> 4;

    // phase 1: cooperative A tile load (64 rows x 128 cols bf16 = 16KB)
    {
        const uint4* src = (const uint4*)(A + (long)m0 * EE);
        uint4* dst = (uint4*)smem;
        for (int i = tid; i < 64 * EE / 8; i += 128) dst[i] = src[i];
    }
    __syncthreads();

    const unsigned short* As = (const unsigned short*)smem;
    FragC acc[2][4];
    #pragma unroll
    for (int nt = 0; nt < 2; ++nt)
      #pragma unroll
      for (int mt = 0; mt < 4; ++mt)
        #pragma unroll
        for (int r = 0; r < 8; ++r) acc[nt][mt].f[r] = 0.f;

    #pragma unroll
    for (int kc = 0; kc < 4; ++kc) {
        FragAB a[4];
        #pragma unroll
        for (int mt = 0; mt < 4; ++mt) {
            const uint4* p = (const uint4*)&As[(mt * 16 + n) * EE + kc * 32 + (half << 4)];
            frag_from_2q(a[mt], p[0], p[1]);   // 16 contiguous ushorts
        }
        #pragma unroll
        for (int nt = 0; nt < 2; ++nt) {
            int ncol = ncolBase + w * 32 + nt * 16 + n;
            FragAB bf;
            const uint4* p = (const uint4*)(Bt + (long)ncol * EE + kc * 32 + (half << 4));
            frag_from_2q(bf, p[0], p[1]);
            #pragma unroll
            for (int mt = 0; mt < 4; ++mt)
                acc[nt][mt].v = __builtin_amdgcn_wmma_f32_16x16x32_bf16(false, a[mt].v, false, bf.v, (short)0, acc[nt][mt].v, false, false);
        }
    }
    __syncthreads();   // A reads done; reuse smem for output staging

    if (mode == 1) {
        unsigned short* Os = (unsigned short*)smem;   // 64x128 bf16
        #pragma unroll
        for (int nt = 0; nt < 2; ++nt) {
            int colT = w * 32 + nt * 16 + n;
            float bsv = bias[ncolBase + colT];
            #pragma unroll
            for (int mt = 0; mt < 4; ++mt)
              #pragma unroll
              for (int r = 0; r < 8; ++r) {
                float val = acc[nt][mt].f[r] + bsv;
                float x3 = val * val * val;
                float g = 0.5f * val * (1.f + tanhf(0.7978845608f * (val + 0.044715f * x3)));
                Os[(mt * 16 + r + (half << 3)) * EE + colT] = f2bf(g);
              }
        }
        __syncthreads();
        const uint4* s4 = (const uint4*)Os;
        for (int i = tid; i < 64 * EE / 8; i += 128) {
            int row = (i * 8) / EE, col = (i * 8) % EE;
            *(uint4*)&outbf[(long)(m0 + row) * N + ncolBase + col] = s4[i];
        }
    } else {
        float* Os = (float*)smem;                      // 64x128 f32
        #pragma unroll
        for (int nt = 0; nt < 2; ++nt) {
            int colT = w * 32 + nt * 16 + n;
            float bsv = bias[ncolBase + colT];
            #pragma unroll
            for (int mt = 0; mt < 4; ++mt)
              #pragma unroll
              for (int r = 0; r < 8; ++r)
                Os[(mt * 16 + r + (half << 3)) * EE + colT] = acc[nt][mt].f[r] + bsv;
        }
        __syncthreads();
        for (int i = tid; i < 64 * EE / 4; i += 128) {
            int row = (i * 4) / EE, col = (i * 4) % EE;
            long idx = (long)(m0 + row) * N + ncolBase + col;
            float4 v = ((const float4*)Os)[i];
            if (mode == 2) {
                float4 rv = *(const float4*)&res[idx];
                v.x += rv.x; v.y += rv.y; v.z += rv.z; v.w += rv.w;
            }
            *(float4*)&outf[idx] = v;
        }
    }
}

extern "C" void kernel_launch(void* const* d_in, const int* in_sizes, int n_in,
                              void* d_out, int out_size, void* d_ws, size_t ws_size,
                              hipStream_t stream) {
    (void)in_sizes; (void)n_in; (void)out_size; (void)ws_size;
    const int*   x       = (const int*)  d_in[0];
    const float* embed   = (const float*)d_in[1];
    const float* ln1_s   = (const float*)d_in[2];
    const float* ln1_b   = (const float*)d_in[3];
    const float* Wg      = (const float*)d_in[4];
    const float* Rg      = (const float*)d_in[5];
    const float* bg      = (const float*)d_in[6];
    const float* gn_s    = (const float*)d_in[7];
    const float* gn_b    = (const float*)d_in[8];
    const float* ln2_s   = (const float*)d_in[9];
    const float* ln2_b   = (const float*)d_in[10];
    const float* W1      = (const float*)d_in[11];
    const float* b1      = (const float*)d_in[12];
    const float* W2      = (const float*)d_in[13];
    const float* b2      = (const float*)d_in[14];
    const float* postn_s = (const float*)d_in[15];
    const float* postn_b = (const float*)d_in[16];
    const float* Wp      = (const float*)d_in[17];
    const float* bp      = (const float*)d_in[18];

    char* ws = (char*)d_ws;
    float* h   = (float*)(ws);                           // 4 MB
    float* yv  = (float*)(ws + (4u  << 20));             // 4 MB (fragment-major)
    float* gx  = (float*)(ws + (8u  << 20));             // 16 MB (fragment-major)
    unsigned short* abf0 = (unsigned short*)(ws + (24u << 20)); // 2 MB
    unsigned short* abf1 = (unsigned short*)(ws + (26u << 20)); // 2 MB
    unsigned short* Wgbf = (unsigned short*)(ws + (28u << 20));
    const int nWgTot = NBLK * 4 * HH * DD * DD;          // 32768
    const int nWgPer = 4 * HH * DD * DD;                 // 16384
    unsigned short* Rgbf = Wgbf + nWgTot;
    unsigned short* W1t  = Rgbf + nWgTot;
    unsigned short* W2t  = W1t + NBLK * EE * EE;
    unsigned short* Wpt  = W2t + NBLK * EE * EE;

    // weight conversions (bf16, transposed N x K where needed)
    k_f32_to_bf16<<<(nWgTot + 255) / 256, 256, 0, stream>>>(Wg, Wgbf, nWgTot);
    k_f32_to_bf16<<<(nWgTot + 255) / 256, 256, 0, stream>>>(Rg, Rgbf, nWgTot);
    for (int blk = 0; blk < NBLK; ++blk) {
        k_f32_to_bf16_tr<<<(EE * EE + 255) / 256, 256, 0, stream>>>(W1 + blk * EE * EE, W1t + blk * EE * EE, EE, EE);
        k_f32_to_bf16_tr<<<(EE * EE + 255) / 256, 256, 0, stream>>>(W2 + blk * EE * EE, W2t + blk * EE * EE, EE, EE);
    }
    k_f32_to_bf16_tr<<<(EE * VV + 255) / 256, 256, 0, stream>>>(Wp, Wpt, EE, VV);

    k_embed<<<ROWS, EE, 0, stream>>>(x, embed, h);

    for (int blk = 0; blk < NBLK; ++blk) {
        k_layernorm_bf<<<ROWS, EE, 0, stream>>>(h, ln1_s + blk * EE, ln1_b + blk * EE, abf0);
        k_gx<<<dim3(SS, NTB), 128, 0, stream>>>(abf0, Wgbf + blk * nWgPer, bg + blk * 4 * HH * DD, gx);
        k_scan<<<dim3(NTB, HH), 32, 0, stream>>>(gx, Rgbf + blk * nWgPer, yv);
        k_hn_res_ln<<<ROWS, EE, 0, stream>>>(yv, gn_s + blk * EE, gn_b + blk * EE,
                                             ln2_s + blk * EE, ln2_b + blk * EE, h, abf0);
        k_gemm128<<<dim3(ROWS / 64, 1), 128, 0, stream>>>(abf0, W1t + blk * EE * EE, b1 + blk * EE,
                                                          nullptr, nullptr, abf1, EE, 1);
        k_gemm128<<<dim3(ROWS / 64, 1), 128, 0, stream>>>(abf1, W2t + blk * EE * EE, b2 + blk * EE,
                                                          h, h, nullptr, EE, 2);
    }
    k_layernorm_bf<<<ROWS, EE, 0, stream>>>(h, postn_s, postn_b, abf0);
    k_gemm128<<<dim3(ROWS / 64, VV / 128), 128, 0, stream>>>(abf0, Wpt, bp,
                                                             nullptr, (float*)d_out, nullptr, VV, 0);
}
